// AttentionLayers_37744172597822
// MI455X (gfx1250) — compile-verified, matrix operand order checked by reading, not run on hardware
//
#include <hip/hip_runtime.h>
#include <hip/hip_bf16.h>
#include <stdint.h>

#define DEVINL __device__ __forceinline__

typedef __attribute__((ext_vector_type(16))) __bf16 bf16x16;
typedef __attribute__((ext_vector_type(8)))  float  f32x8;
typedef __attribute__((ext_vector_type(4)))  unsigned u32x4;

namespace {
constexpr int Bsz  = 2;
constexpr int Nseq = 2048;
constexpr int Dim  = 512;
constexpr int H    = 8;
constexpr int DHd  = 64;
constexpr int Mem  = 16;
constexpr int J    = Mem + Nseq;     // 2064
constexpr int JP   = 2080;           // padded vT row stride (multiple of 32)
constexpr int JT16 = J / 16;         // 129 column tiles
constexpr int JT32 = (JT16 + 1) / 2; // 65 (pairs of 16-tiles -> K=32 PV steps)
constexpr int Rows = Bsz * Nseq;     // 4096
constexpr int KROW = 72;             // LDS k-stage row stride in shorts (144 B)
}

// ---------- helpers ----------

DEVINL unsigned short f2bf(float x) {
  unsigned u = __float_as_uint(x);
  u = (u + 0x7FFFu + ((u >> 16) & 1u)) >> 16;   // round-to-nearest-even
  return (unsigned short)u;
}

union FragU { bf16x16 v; u32x4 q[2]; unsigned u[8]; };

// Load a 16x32 bf16 A (or 32x16 B) fragment. rowp points at (lane-row, k0) in a
// row-major bf16 buffer whose K dimension is contiguous per lane.
// Layout per ISA 7.12.2: lanes 0-15 hold K = {0..7, 16..23}, lanes 16-31 hold
// K = {8..15, 24..31}; dword r<4 holds K=2r,2r+1 (+8*half), r>=4 holds 16+...
DEVINL bf16x16 load_frag(const unsigned short* rowp, int half) {
  const u32x4* p = (const u32x4*)rowp;
  FragU f;
  f.q[0] = p[half];       // dwords 4h..4h+3   -> K 8h..8h+7
  f.q[1] = p[2 + half];   // dwords 8+4h..     -> K 16+8h..23+8h
  return f.v;
}

// Build a bf16 A fragment from a 16x32 fp32 tile in LDS (post-mixed P tile).
DEVINL bf16x16 frag_from_lds(const float* tile /*16x32*/, int lanelo, int half) {
  const float* row = tile + lanelo * 32;
  FragU f;
#pragma unroll
  for (int r = 0; r < 8; ++r) {
    int k0 = (r < 4) ? (2 * r + 8 * half) : (16 + 2 * (r - 4) + 8 * half);
    unsigned lo = f2bf(row[k0]);
    unsigned hi = f2bf(row[k0 + 1]);
    f.u[r] = lo | (hi << 16);
  }
  return f.v;
}

DEVINL f32x8 wmma_bf16(bf16x16 a, bf16x16 b, f32x8 c) {
  return __builtin_amdgcn_wmma_f32_16x16x32_bf16(
      /*neg_a=*/false, a, /*neg_b=*/false, b,
      /*c_mod=*/(short)0, c, /*reuse_a=*/false, /*reuse_b=*/false);
}

// 16-lane reductions (masks < 16 stay inside each wave32 half; matches the
// WMMA C layout where one output row is striped across 16 lanes).
DEVINL float rmax16(float v) {
#pragma unroll
  for (int m = 1; m < 16; m <<= 1) v = fmaxf(v, __shfl_xor(v, m, 32));
  return v;
}
DEVINL float rsum16(float v) {
#pragma unroll
  for (int m = 1; m < 16; m <<= 1) v += __shfl_xor(v, m, 32);
  return v;
}

// Async DMA one contiguous 2KB global K tile (16 rows x 128B) into LDS with
// rows padded to 144B (bank-conflict-free ds_load_b128 fragment reads).
// Tracked on ASYNCcnt; 4 x global_load_async_to_lds_b128 per lane-set.
DEVINL void stage_tile_async(const unsigned short* gtile, unsigned ldsbase, int lane) {
#pragma unroll
  for (int i = 0; i < 4; ++i) {
    int off = lane * 64 + i * 16;   // byte offset within the 2KB tile
    int row = off >> 7;
    int col = off & 127;
    unsigned laddr = ldsbase + (unsigned)(row * (KROW * 2) + col);
    unsigned long long gaddr = (unsigned long long)((const char*)gtile + off);
    asm volatile("global_load_async_to_lds_b128 %0, %1, off"
                 :: "v"(laddr), "v"(gaddr)
                 : "memory");
  }
}

// ---------- conversion kernels ----------

__global__ void cvt_f32_bf16(const float* __restrict__ src,
                             unsigned short* __restrict__ dst, int n) {
  int i = blockIdx.x * blockDim.x + threadIdx.x;
  if (i < n) dst[i] = f2bf(src[i]);
}

// 512x512 fp32 -> bf16, transposed so B-fragments read K contiguously per lane.
__global__ void cvt_transpose512(const float* __restrict__ src,
                                 unsigned short* __restrict__ dst) {
  int i = blockIdx.x * blockDim.x + threadIdx.x;
  if (i < 512 * 512) {
    int k = i >> 9, n = i & 511;
    dst[n * 512 + k] = f2bf(src[i]);
  }
}

// memory k/v -> prepended rows of kmem [b,h,J,dh] and vT [b,h,dh,JP]
__global__ void cvt_mem(const float* __restrict__ mk, const float* __restrict__ mv,
                        unsigned short* __restrict__ kb, unsigned short* __restrict__ vT) {
  int i = blockIdx.x * blockDim.x + threadIdx.x;
  if (i < H * Mem * DHd) {
    int h = i / (Mem * DHd);
    int j = (i / DHd) % Mem;
    int d = i % DHd;
    unsigned short kv = f2bf(mk[i]);
    unsigned short vv = f2bf(mv[i]);
#pragma unroll
    for (int b = 0; b < Bsz; ++b) {
      kb[((size_t)(b * H + h) * J + j) * DHd + d] = kv;
      vT[((size_t)(b * H + h) * DHd + d) * JP + j] = vv;
    }
  }
}

// zero the vT tail pad (j in [J, JP)) so the final K=32 PV step reads finite 0s
__global__ void pad_vT(unsigned short* __restrict__ vT) {
  int i = blockIdx.x * blockDim.x + threadIdx.x;
  if (i < Bsz * H * DHd * (JP - J)) {
    int row = i / (JP - J);
    int jp = i % (JP - J);
    vT[(size_t)row * JP + J + jp] = 0;
  }
}

// ---------- GEMM: QKV projections ----------
// C[4096 x 512] = xb[4096 x 512] @ W[512 x 512]; blockIdx.z picks q/k/v and
// the scattered (split-heads) output layout.
__global__ void __launch_bounds__(256) qkv_gemm(
    const unsigned short* __restrict__ xb,
    const unsigned short* __restrict__ wqT,
    const unsigned short* __restrict__ wkT,
    const unsigned short* __restrict__ wvT,
    unsigned short* __restrict__ qb,
    unsigned short* __restrict__ kb,
    unsigned short* __restrict__ vT) {
  const int lane = threadIdx.x & 31;
  const int lanelo = lane & 15;
  const int half = (lane >> 4) & 1;
  const int wv = threadIdx.x >> 5;              // 8 waves: 2x4 tile grid
  const int M0 = blockIdx.x * 32 + (wv >> 2) * 16;
  const int N0 = blockIdx.y * 64 + (wv & 3) * 16;
  const unsigned short* wT =
      (blockIdx.z == 0) ? wqT : (blockIdx.z == 1) ? wkT : wvT;

  const unsigned short* arow = xb + (size_t)(M0 + lanelo) * Dim;
  const unsigned short* brow = wT + (size_t)(N0 + lanelo) * Dim;
  f32x8 acc = {};
#pragma unroll 4
  for (int kk = 0; kk < Dim; kk += 32) {
    bf16x16 a = load_frag(arow + kk, half);
    bf16x16 b = load_frag(brow + kk, half);
    acc = wmma_bf16(a, b, acc);
  }
#pragma unroll
  for (int r = 0; r < 8; ++r) {
    int M = M0 + r + 8 * half;
    int col = N0 + lanelo;
    int bb = M >> 11, nn = M & (Nseq - 1);
    int hh = col >> 6, dd = col & 63;
    unsigned short v = f2bf(acc[r]);
    if (blockIdx.z == 0)
      qb[((size_t)(bb * H + hh) * Nseq + nn) * DHd + dd] = v;
    else if (blockIdx.z == 1)
      kb[((size_t)(bb * H + hh) * J + (Mem + nn)) * DHd + dd] = v;
    else
      vT[((size_t)(bb * H + hh) * DHd + dd) * JP + (Mem + nn)] = v;
  }
}

// ---------- fused talking-heads flash attention ----------
// Grid: (Nseq/16, Bsz). 8 waves = 8 heads per workgroup, 16 query rows.
// Pass 1: streaming row max + sum; K tiles double-buffered in LDS via
//         global_load_async_to_lds_b128 (ASYNCcnt), one tile ahead.
// Pass 2: recompute scores, normalize, post-mix P through LDS, P@V via WMMA,
//         with global_prefetch of the next K/V tiles.
__global__ void __launch_bounds__(256) attn_fused(
    const unsigned short* __restrict__ qb,
    const unsigned short* __restrict__ kb,
    const unsigned short* __restrict__ vT,
    const float* __restrict__ pre, const float* __restrict__ post,
    const float* __restrict__ hsp,
    unsigned short* __restrict__ aout) {
  __shared__ float st[H][16][16];   // cross-head 16x16 exchange tiles
  __shared__ float pm[H][16][32];   // post-mixed P, K=32 staging for PV
  __shared__ __align__(16) unsigned short kstage[H][2][16 * KROW]; // async K tiles

  const int lane = threadIdx.x & 31;
  const int lanelo = lane & 15;
  const int half = (lane >> 4) & 1;
  const int w = threadIdx.x >> 5;   // this wave's head index
  const int rt = blockIdx.x;        // 16-row tile
  const int b = blockIdx.y;
  const int bh = b * H + w;

  float prew[H], postw[H];
#pragma unroll
  for (int h = 0; h < H; ++h) {
    prew[h] = pre[h * H + w];
    postw[h] = post[h * H + w];
  }
  const float slope = exp2f(-(float)(w + 1));  // alibi slopes for h=8
  const float hs = hsp[w];
  const float scale = 0.125f;                  // dh^-0.5

  const unsigned short* qrow =
      qb + ((size_t)bh * Nseq + rt * 16 + lanelo) * DHd;
  const bf16x16 qa0 = load_frag(qrow, half);
  const bf16x16 qa1 = load_frag(qrow + 32, half);

  float m_r[8], l_r[8];
#pragma unroll
  for (int r = 0; r < 8; ++r) { m_r[r] = -3.0e38f; l_r[r] = 0.0f; }

  unsigned short* kst[2] = { &kstage[w][0][0], &kstage[w][1][0] };
  const unsigned kbase[2] = { (unsigned)(unsigned long long)kst[0],
                              (unsigned)(unsigned long long)kst[1] };

  // ---- pass 1: row max + normalizer ----
  stage_tile_async(kb + (size_t)bh * J * DHd, kbase[0], lane);  // tile jt=0
  int buf = 0;
  for (int jt = 0; jt < JT16; ++jt) {
    if (jt + 1 < JT16) {
      stage_tile_async(kb + ((size_t)bh * J + (size_t)(jt + 1) * 16) * DHd,
                       kbase[buf ^ 1], lane);
      asm volatile("s_wait_asynccnt 4" ::: "memory");  // current tile landed
    } else {
      asm volatile("s_wait_asynccnt 0" ::: "memory");
    }
    const unsigned short* krow = kst[buf] + lanelo * KROW;
    f32x8 c = {};
    c = wmma_bf16(qa0, load_frag(krow, half), c);
    c = wmma_bf16(qa1, load_frag(krow + 32, half), c);
    __syncthreads();
#pragma unroll
    for (int r = 0; r < 8; ++r) st[w][r + 8 * half][lanelo] = c[r] * scale;
    __syncthreads();
#pragma unroll
    for (int r = 0; r < 8; ++r) {
      int M = r + 8 * half;
      int row = rt * 16 + M;
      int col = jt * 16 + lanelo;
      float s = 0.0f;
#pragma unroll
      for (int h = 0; h < H; ++h) s += st[h][M][lanelo] * prew[h];
      s += (float)col * slope;
      bool msk = (col - Mem) > row;
      s = msk ? -3.0e38f : s;
      float tm = rmax16(s);
      float mn = fmaxf(m_r[r], tm);
      float e = msk ? 0.0f : __expf(s - mn);
      float es = rsum16(e);
      l_r[r] = l_r[r] * __expf(m_r[r] - mn) + es;
      m_r[r] = mn;
    }
    buf ^= 1;
  }
  float il[8];
#pragma unroll
  for (int r = 0; r < 8; ++r) il[r] = 1.0f / l_r[r];

  // ---- pass 2: normalized scores, post-mix, P@V ----
  f32x8 acc[4] = {};
  for (int jt2 = 0; jt2 < JT32; ++jt2) {
#pragma unroll
    for (int sub = 0; sub < 2; ++sub) {
      const int jt = jt2 * 2 + sub;
      const bool valid = (jt < JT16);  // block-uniform
      f32x8 c = {};
      if (valid) {
        const unsigned short* krow =
            kb + ((size_t)bh * J + jt * 16 + lanelo) * DHd;
        c = wmma_bf16(qa0, load_frag(krow, half), c);
        c = wmma_bf16(qa1, load_frag(krow + 32, half), c);
        if (jt + 2 < JT16)  // pull next K tile toward the WGP early
          __builtin_prefetch(kb + ((size_t)bh * J + (size_t)(jt + 2) * 16 + lanelo) * DHd, 0, 0);
      }
      __syncthreads();
      if (valid) {
#pragma unroll
        for (int r = 0; r < 8; ++r) st[w][r + 8 * half][lanelo] = c[r] * scale;
      }
      __syncthreads();
      float p_r[8];
#pragma unroll
      for (int r = 0; r < 8; ++r) {
        p_r[r] = 0.0f;
        if (valid) {
          int M = r + 8 * half;
          int row = rt * 16 + M;
          int col = jt * 16 + lanelo;
          float s = 0.0f;
#pragma unroll
          for (int h = 0; h < H; ++h) s += st[h][M][lanelo] * prew[h];
          s += (float)col * slope;
          bool msk = (col - Mem) > row;
          p_r[r] = msk ? 0.0f : __expf(s - m_r[r]) * il[r];
        }
      }
      __syncthreads();
#pragma unroll
      for (int r = 0; r < 8; ++r) st[w][r + 8 * half][lanelo] = p_r[r];
      __syncthreads();
#pragma unroll
      for (int r = 0; r < 8; ++r) {
        int M = r + 8 * half;
        float pv = 0.0f;
#pragma unroll
        for (int h = 0; h < H; ++h) pv += st[h][M][lanelo] * postw[h];
        pm[w][M][sub * 16 + lanelo] = pv;
      }
    }
    __syncthreads();
    if (jt2 + 1 < JT32) {  // prefetch next V rows
#pragma unroll
      for (int c0 = 0; c0 < 4; ++c0)
        __builtin_prefetch(
            vT + ((size_t)bh * DHd + c0 * 16 + lanelo) * JP + (size_t)(jt2 + 1) * 32, 0, 0);
    }
    bf16x16 pa = frag_from_lds(&pm[w][0][0], lanelo, half);
#pragma unroll
    for (int c0 = 0; c0 < 4; ++c0) {
      const unsigned short* vrow =
          vT + ((size_t)bh * DHd + c0 * 16 + lanelo) * JP + (size_t)jt2 * 32;
      acc[c0] = wmma_bf16(pa, load_frag(vrow, half), acc[c0]);
    }
  }

  // head-scale + write [b, n, h*dh] bf16
#pragma unroll
  for (int c0 = 0; c0 < 4; ++c0) {
#pragma unroll
    for (int r = 0; r < 8; ++r) {
      int M = r + 8 * half;
      size_t row = (size_t)b * Nseq + rt * 16 + M;
      aout[row * (H * DHd) + w * DHd + c0 * 16 + lanelo] = f2bf(acc[c0][r] * hs);
    }
  }
}

// ---------- output projection GEMM (+bias, fp32 out) ----------
__global__ void __launch_bounds__(256) out_gemm(
    const unsigned short* __restrict__ aout,
    const unsigned short* __restrict__ woT,
    const float* __restrict__ bout,
    float* __restrict__ out) {
  const int lane = threadIdx.x & 31;
  const int lanelo = lane & 15;
  const int half = (lane >> 4) & 1;
  const int wv = threadIdx.x >> 5;
  const int M0 = blockIdx.x * 32 + (wv >> 2) * 16;
  const int N0 = blockIdx.y * 64 + (wv & 3) * 16;

  const unsigned short* arow = aout + (size_t)(M0 + lanelo) * Dim;
  const unsigned short* brow = woT + (size_t)(N0 + lanelo) * Dim;
  f32x8 acc = {};
#pragma unroll 4
  for (int kk = 0; kk < Dim; kk += 32) {
    bf16x16 a = load_frag(arow + kk, half);
    bf16x16 b = load_frag(brow + kk, half);
    acc = wmma_bf16(a, b, acc);
  }
#pragma unroll
  for (int r = 0; r < 8; ++r) {
    int M = M0 + r + 8 * half;
    int col = N0 + lanelo;
    out[(size_t)M * Dim + col] = acc[r] + bout[col];
  }
}

// ---------- launcher ----------

extern "C" void kernel_launch(void* const* d_in, const int* in_sizes, int n_in,
                              void* d_out, int out_size, void* d_ws, size_t ws_size,
                              hipStream_t stream) {
  const float* x     = (const float*)d_in[0];
  const float* Wq    = (const float*)d_in[1];
  const float* Wk    = (const float*)d_in[2];
  const float* Wv    = (const float*)d_in[3];
  const float* mem_k = (const float*)d_in[4];
  const float* mem_v = (const float*)d_in[5];
  const float* pre   = (const float*)d_in[6];
  const float* post  = (const float*)d_in[7];
  const float* hsp   = (const float*)d_in[8];
  const float* Wout  = (const float*)d_in[9];
  const float* bout  = (const float*)d_in[10];
  float* out = (float*)d_out;

  char* ws = (char*)d_ws;
  size_t off = 0;
  auto take = [&](size_t bytes) -> char* {
    char* p = ws + off;
    off = (off + bytes + 255) & ~(size_t)255;
    return p;
  };
  unsigned short* xb   = (unsigned short*)take((size_t)Rows * Dim * 2);
  unsigned short* wqT  = (unsigned short*)take((size_t)Dim * Dim * 2);
  unsigned short* wkT  = (unsigned short*)take((size_t)Dim * Dim * 2);
  unsigned short* wvT  = (unsigned short*)take((size_t)Dim * Dim * 2);
  unsigned short* woT  = (unsigned short*)take((size_t)Dim * Dim * 2);
  unsigned short* qb   = (unsigned short*)take((size_t)Bsz * H * Nseq * DHd * 2);
  unsigned short* kbuf = (unsigned short*)take((size_t)Bsz * H * J * DHd * 2);
  unsigned short* vT   = (unsigned short*)take((size_t)Bsz * H * DHd * JP * 2);
  unsigned short* aout = (unsigned short*)take((size_t)Rows * Dim * 2);
  (void)in_sizes; (void)n_in; (void)out_size; (void)ws_size;

  const int thr = 256;
  cvt_f32_bf16<<<(Rows * Dim + thr - 1) / thr, thr, 0, stream>>>(x, xb, Rows * Dim);
  cvt_transpose512<<<(Dim * Dim + thr - 1) / thr, thr, 0, stream>>>(Wq, wqT);
  cvt_transpose512<<<(Dim * Dim + thr - 1) / thr, thr, 0, stream>>>(Wk, wkT);
  cvt_transpose512<<<(Dim * Dim + thr - 1) / thr, thr, 0, stream>>>(Wv, wvT);
  cvt_transpose512<<<(Dim * Dim + thr - 1) / thr, thr, 0, stream>>>(Wout, woT);
  cvt_mem<<<(H * Mem * DHd + thr - 1) / thr, thr, 0, stream>>>(mem_k, mem_v, kbuf, vT);
  pad_vT<<<(Bsz * H * DHd * (JP - J) + thr - 1) / thr, thr, 0, stream>>>(vT);

  qkv_gemm<<<dim3(Rows / 32, Dim / 64, 3), thr, 0, stream>>>(
      xb, wqT, wkT, wvT, qb, kbuf, vT);

  attn_fused<<<dim3(Nseq / 16, Bsz), thr, 0, stream>>>(
      qb, kbuf, vT, pre, post, hsp, aout);

  out_gemm<<<dim3(Rows / 32, Dim / 64), thr, 0, stream>>>(aout, woT, bout, out);
}
